// GAT_2869038154141
// MI455X (gfx1250) — compile-verified, matrix operand order checked by reading, not run on hardware
//
#include <hip/hip_runtime.h>
#include <hip/hip_bf16.h>

#define N      8192
#define NFEAT  512
#define NHID   64
#define NHEADS 4
#define NCLASS 16
#define KCAT   (NHEADS * NHID)   // 256
#define ALPHA      0.2f
#define OUT_SLOPE  0.01f
#define LOG2E      1.4426950408889634f

typedef __attribute__((ext_vector_type(16))) _Float16 v16h;
typedef __attribute__((ext_vector_type(8)))  _Float16 h8;
typedef __attribute__((ext_vector_type(8)))  float    v8f;

union F16x16 { v16h v; h8 h[2]; };

static __device__ __forceinline__ h8 ldh8(const _Float16* p) { return *(const h8*)p; }

static __device__ __forceinline__ v8f wmma_f16(v16h a, v16h b, v8f c) {
  // D = A(16x32 f16) * B(32x16 f16) + C(16x16 f32)
  return __builtin_amdgcn_wmma_f32_16x16x32_f16(false, a, false, b, (short)0, c, false, false);
}

static __device__ __forceinline__ float lrelu(float x, float s) {
  return x > 0.f ? x : s * x;
}

// packed score -> probability: p = mask * exp2( (0.6*u + 0.4*|u|)*log2e - m*log2e )
// all ops v_pk_* / v_exp_f16
static __device__ __forceinline__ h8 p_octet(h8 f2v, _Float16 f1v, _Float16 mneg, h8 maskv) {
  const _Float16 C6 = (_Float16)(0.6f * LOG2E);
  const _Float16 C4 = (_Float16)(0.4f * LOG2E);
  h8 u  = f2v + f1v;
  h8 au = __builtin_elementwise_max(u, -u);
  h8 t  = au * C4 + mneg;
  h8 g  = u * C6 + t;
  h8 e  = __builtin_elementwise_exp2(g);
  return e * maskv;
}

// ---------------- layer-1 attention (32 rows/block, 4 heads x 2 phases) ----------------
__global__ void __launch_bounds__(256, 1)
k_att1(const unsigned* __restrict__ bm,
       const _Float16* __restrict__ H16T,
       const float* __restrict__ f1, const _Float16* __restrict__ f2h16,
       const float* __restrict__ gmax,
       _Float16* __restrict__ xcat16) {
  __shared__ float accs[32 * KCAT];        // 32 KB
  __shared__ float dls[NHEADS * 32];
  __shared__ uint4 lutm[256];              // mask byte -> 8 x f16 {0,1}
  const int row0 = blockIdx.x * 32;
  const int tid  = threadIdx.x;
  const int w = tid >> 5, lane = tid & 31, lr = lane & 15, ks = lane >> 4;
  const int h = w >> 1, g = w & 1;

  {
    union { _Float16 hh[8]; uint4 u; } e;
    for (int b = 0; b < 8; b++) e.hh[b] = (_Float16)((tid >> b) & 1);
    lutm[tid] = e.u;
  }
  for (int i = tid; i < 32 * KCAT; i += 256) accs[i] = 0.f;
  if (tid < NHEADS * 32) dls[tid] = 0.f;
  __syncthreads();

  _Float16 f1h[2], mneg[2];
  for (int rt = 0; rt < 2; rt++) {
    float f1v = f1[h * N + row0 + rt * 16 + lr];
    float m   = lrelu(f1v + gmax[h], ALPHA);   // >= masked row max (lrelu monotone)
    f1h[rt]  = (_Float16)f1v;
    mneg[rt] = (_Float16)(-m * LOG2E);
  }
  const _Float16* f2h  = f2h16 + (size_t)h * N;
  const unsigned* bmr0 = bm + (size_t)(row0 + lr) * (N / 32);
  const unsigned* bmr1 = bm + (size_t)(row0 + 16 + lr) * (N / 32);
  const _Float16* Bh   = H16T + (size_t)h * NHID * N + (size_t)lr * N;

  v8f acc[2][4] = {};
  v8f accd[2] = {};
  F16x16 ones;
  for (int i = 0; i < 8; i++) { ones.h[0][i] = (_Float16)1.f; ones.h[1][i] = (_Float16)1.f; }

  for (int c = g; c < N / 32; c += 2) {
    const int j0 = c * 32;
    unsigned mwv[2] = { bmr0[c], bmr1[c] };
    h8 fa = ldh8(f2h + j0 + ks * 8);
    h8 fb = ldh8(f2h + j0 + 16 + ks * 8);
    F16x16 pa[2];
    for (int rt = 0; rt < 2; rt++) {
      unsigned mw = mwv[rt];
      h8 ma = *(const h8*)&lutm[(mw >> (8 * ks)) & 255u];
      h8 mb = *(const h8*)&lutm[(mw >> (16 + 8 * ks)) & 255u];
      pa[rt].h[0] = p_octet(fa, f1h[rt], mneg[rt], ma);
      pa[rt].h[1] = p_octet(fb, f1h[rt], mneg[rt], mb);
    }
    for (int t = 0; t < 4; t++) {
      const _Float16* bp = Bh + (size_t)t * 16 * N + j0;
      F16x16 bb;
      bb.h[0] = ldh8(bp + ks * 8);
      bb.h[1] = ldh8(bp + 16 + ks * 8);
      acc[0][t] = wmma_f16(pa[0].v, bb.v, acc[0][t]);
      acc[1][t] = wmma_f16(pa[1].v, bb.v, acc[1][t]);
    }
    accd[0] = wmma_f16(pa[0].v, ones.v, accd[0]);   // row sums (softmax denominator)
    accd[1] = wmma_f16(pa[1].v, ones.v, accd[1]);
  }
  for (int rt = 0; rt < 2; rt++) {
    if (lr == 0)
      for (int v = 0; v < 8; v++)
        atomicAdd(&dls[h * 32 + rt * 16 + v + 8 * ks], accd[rt][v]);
    for (int t = 0; t < 4; t++)
      for (int v = 0; v < 8; v++)
        atomicAdd(&accs[(rt * 16 + v + 8 * ks) * KCAT + h * NHID + t * 16 + lr], acc[rt][t][v]);
  }
  __syncthreads();
  for (int idx = tid; idx < 32 * KCAT; idx += 256) {
    int r = idx >> 8, cc = idx & 255;
    int hh = cc >> 6;
    float d = dls[hh * 32 + r];
    float v = (d > 0.f) ? accs[idx] / d : 0.f;
    xcat16[(size_t)(row0 + r) * KCAT + cc] = (_Float16)lrelu(v, OUT_SLOPE);
  }
}

// ---------------- layer-2 attention -> output (32 rows/block) ----------------
__global__ void __launch_bounds__(256, 1)
k_att2(const unsigned* __restrict__ bm,
       const _Float16* __restrict__ H2T,
       const float* __restrict__ f1o, const _Float16* __restrict__ f2o16,
       const float* __restrict__ gmaxo,
       float* __restrict__ out) {
  __shared__ float accs[32 * NCLASS];
  __shared__ float dls[32];
  __shared__ uint4 lutm[256];
  const int row0 = blockIdx.x * 32;
  const int tid  = threadIdx.x;
  const int w = tid >> 5, lane = tid & 31, lr = lane & 15, ks = lane >> 4;

  {
    union { _Float16 hh[8]; uint4 u; } e;
    for (int b = 0; b < 8; b++) e.hh[b] = (_Float16)((tid >> b) & 1);
    lutm[tid] = e.u;
  }
  for (int i = tid; i < 32 * NCLASS; i += 256) accs[i] = 0.f;
  if (tid < 32) dls[tid] = 0.f;
  __syncthreads();

  _Float16 f1h[2], mneg[2];
  for (int rt = 0; rt < 2; rt++) {
    float f1v = f1o[row0 + rt * 16 + lr];
    float m   = lrelu(f1v + gmaxo[0], ALPHA);
    f1h[rt]  = (_Float16)f1v;
    mneg[rt] = (_Float16)(-m * LOG2E);
  }
  const unsigned* bmr0 = bm + (size_t)(row0 + lr) * (N / 32);
  const unsigned* bmr1 = bm + (size_t)(row0 + 16 + lr) * (N / 32);
  const _Float16* Brow = H2T + (size_t)lr * N;

  v8f acc[2] = {};
  v8f accd[2] = {};
  F16x16 ones;
  for (int i = 0; i < 8; i++) { ones.h[0][i] = (_Float16)1.f; ones.h[1][i] = (_Float16)1.f; }

  for (int c = w; c < N / 32; c += 8) {
    const int j0 = c * 32;
    unsigned mwv[2] = { bmr0[c], bmr1[c] };
    h8 fa = ldh8(f2o16 + j0 + ks * 8);
    h8 fb = ldh8(f2o16 + j0 + 16 + ks * 8);
    F16x16 pa[2];
    for (int rt = 0; rt < 2; rt++) {
      unsigned mw = mwv[rt];
      h8 ma = *(const h8*)&lutm[(mw >> (8 * ks)) & 255u];
      h8 mb = *(const h8*)&lutm[(mw >> (16 + 8 * ks)) & 255u];
      pa[rt].h[0] = p_octet(fa, f1h[rt], mneg[rt], ma);
      pa[rt].h[1] = p_octet(fb, f1h[rt], mneg[rt], mb);
    }
    F16x16 bb;
    bb.h[0] = ldh8(Brow + j0 + ks * 8);
    bb.h[1] = ldh8(Brow + j0 + 16 + ks * 8);
    acc[0]  = wmma_f16(pa[0].v, bb.v, acc[0]);
    acc[1]  = wmma_f16(pa[1].v, bb.v, acc[1]);
    accd[0] = wmma_f16(pa[0].v, ones.v, accd[0]);
    accd[1] = wmma_f16(pa[1].v, ones.v, accd[1]);
  }
  for (int rt = 0; rt < 2; rt++) {
    if (lr == 0)
      for (int v = 0; v < 8; v++)
        atomicAdd(&dls[rt * 16 + v + 8 * ks], accd[rt][v]);
    for (int v = 0; v < 8; v++)
      atomicAdd(&accs[(rt * 16 + v + 8 * ks) * NCLASS + lr], acc[rt][v]);
  }
  __syncthreads();
  for (int idx = tid; idx < 32 * NCLASS; idx += 256) {
    int r = idx >> 4, n = idx & 15;
    float d = dls[r];
    out[(size_t)(row0 + r) * NCLASS + n] = (d > 0.f) ? accs[r * NCLASS + n] / d : 0.f;
  }
}

// ---------------- conversion kernels ----------------
__global__ void k_cvt_x(const float* __restrict__ x, _Float16* __restrict__ x16, int n) {
  int i = blockIdx.x * 256 + threadIdx.x;
  if (i < n) x16[i] = (_Float16)__builtin_nontemporal_load(x + i);
}

// Wh[h][k][n] (4,512,64) -> W16T[h*64+n][k]
__global__ void k_cvt_wh(const float* __restrict__ Wh, _Float16* __restrict__ W16T) {
  int idx = blockIdx.x * 256 + threadIdx.x;
  if (idx < NHEADS * NFEAT * NHID) {
    int n = idx % NHID;
    int k = (idx / NHID) % NFEAT;
    int h = idx / (NHID * NFEAT);
    W16T[(h * NHID + n) * NFEAT + k] = (_Float16)Wh[idx];
  }
}

// Wo[k][n] (256,16) -> Wo16T[n][k]
__global__ void k_cvt_wo(const float* __restrict__ Wo, _Float16* __restrict__ Wo16T) {
  int idx = blockIdx.x * 256 + threadIdx.x;
  if (idx < KCAT * NCLASS) {
    int n = idx % NCLASS;
    int k = idx / NCLASS;
    Wo16T[n * KCAT + k] = (_Float16)Wo[idx];
  }
}

// ---------------- adj -> bitmask (stream 268MB once with NT hint, write 8MB) ----------------
__global__ void k_pack(const int* __restrict__ adj, unsigned* __restrict__ bm) {
  const int col = blockIdx.x * 256 + threadIdx.x;
  const int row = blockIdx.y;
  const int pred = __builtin_nontemporal_load(adj + (size_t)row * N + col) > 0;
  unsigned mask = __builtin_amdgcn_ballot_w32(pred);
  if ((threadIdx.x & 31) == 0) bm[row * (N / 32) + (col >> 5)] = mask;
}

// ---------------- H = x @ Wh (all heads), f1/f2, H16T ----------------
__global__ void k_gemm1(const _Float16* __restrict__ x16,
                        const _Float16* __restrict__ W16T,
                        const float* __restrict__ ah,
                        _Float16* __restrict__ H16T,
                        float* __restrict__ f1, float* __restrict__ f2,
                        _Float16* __restrict__ f2h16) {
  __shared__ float Hs[16 * KCAT];
  const int row0 = blockIdx.x * 16;
  const int tid  = threadIdx.x;
  const int w = tid >> 5, lane = tid & 31, lr = lane & 15, ks = lane >> 4;

  v8f acc0 = {}, acc1 = {};
  const _Float16* arow = x16  + (size_t)(row0 + lr) * NFEAT;
  const _Float16* b0   = W16T + (size_t)(w * 32 + lr) * NFEAT;
  const _Float16* b1   = W16T + (size_t)(w * 32 + 16 + lr) * NFEAT;
  for (int kk = 0; kk < NFEAT; kk += 32) {
    F16x16 a, bA, bB;
    a.h[0]  = ldh8(arow + kk + ks * 8);
    a.h[1]  = ldh8(arow + kk + 16 + ks * 8);
    bA.h[0] = ldh8(b0 + kk + ks * 8);
    bA.h[1] = ldh8(b0 + kk + 16 + ks * 8);
    bB.h[0] = ldh8(b1 + kk + ks * 8);
    bB.h[1] = ldh8(b1 + kk + 16 + ks * 8);
    acc0 = wmma_f16(a.v, bA.v, acc0);
    acc1 = wmma_f16(a.v, bB.v, acc1);
  }
  for (int v = 0; v < 8; v++) {
    int r = v + 8 * ks;
    Hs[r * KCAT + w * 32 + lr]      = acc0[v];
    Hs[r * KCAT + w * 32 + 16 + lr] = acc1[v];
  }
  __syncthreads();
  if (tid < NHEADS * 16) {
    int h = tid >> 4, r = tid & 15;
    float s1 = 0.f, s2 = 0.f;
    for (int k = 0; k < NHID; k++) {
      float v = Hs[r * KCAT + h * NHID + k];
      s1 += v * ah[h * 2 * NHID + k];
      s2 += v * ah[h * 2 * NHID + NHID + k];
    }
    f1[h * N + row0 + r]    = s1;
    f2[h * N + row0 + r]    = s2;
    f2h16[h * N + row0 + r] = (_Float16)s2;
  }
  for (int idx = tid; idx < 16 * KCAT; idx += 256) {
    int r = idx >> 8;
    int c = idx & 255;
    H16T[(size_t)c * N + row0 + r] = (_Float16)Hs[r * KCAT + c];
  }
}

// ---------------- global max per head ----------------
__global__ void k_rowmax(const float* __restrict__ v, float* __restrict__ out) {
  __shared__ float red[256];
  const float* p = v + (size_t)blockIdx.x * N;
  float m = -1e30f;
  for (int i = threadIdx.x; i < N; i += 256) m = fmaxf(m, p[i]);
  red[threadIdx.x] = m;
  __syncthreads();
  for (int s = 128; s > 0; s >>= 1) {
    if (threadIdx.x < s) red[threadIdx.x] = fmaxf(red[threadIdx.x], red[threadIdx.x + s]);
    __syncthreads();
  }
  if (threadIdx.x == 0) out[blockIdx.x] = red[0];
}

// ---------------- h2 = xcat @ Wo, f1o/f2o, H2T ----------------
__global__ void k_gemm2(const _Float16* __restrict__ xcat16,
                        const _Float16* __restrict__ Wo16T,
                        const float* __restrict__ ao,
                        _Float16* __restrict__ H2T,
                        float* __restrict__ f1o, float* __restrict__ f2o,
                        _Float16* __restrict__ f2o16) {
  __shared__ float Hs[16 * 16];
  const int row0 = blockIdx.x * 16;
  const int lane = threadIdx.x & 31, lr = lane & 15, ks = lane >> 4;
  v8f acc = {};
  const _Float16* arow = xcat16 + (size_t)(row0 + lr) * KCAT;
  const _Float16* brow = Wo16T + (size_t)lr * KCAT;
  for (int kk = 0; kk < KCAT; kk += 32) {
    F16x16 a, b;
    a.h[0] = ldh8(arow + kk + ks * 8);
    a.h[1] = ldh8(arow + kk + 16 + ks * 8);
    b.h[0] = ldh8(brow + kk + ks * 8);
    b.h[1] = ldh8(brow + kk + 16 + ks * 8);
    acc = wmma_f16(a.v, b.v, acc);
  }
  for (int v = 0; v < 8; v++) Hs[(v + 8 * ks) * 16 + lr] = acc[v];
  __syncthreads();
  if (threadIdx.x < 16) {
    int r = threadIdx.x;
    float s1 = 0.f, s2 = 0.f;
    for (int n = 0; n < NCLASS; n++) {
      float v = Hs[r * 16 + n];
      s1 += v * ao[n];
      s2 += v * ao[NCLASS + n];
    }
    f1o[row0 + r]   = s1;
    f2o[row0 + r]   = s2;
    f2o16[row0 + r] = (_Float16)s2;
  }
  for (int idx = threadIdx.x; idx < 256; idx += 32) {
    int r = idx >> 4, n = idx & 15;
    H2T[(size_t)n * N + row0 + r] = (_Float16)Hs[r * 16 + n];
  }
}

// ---------------- host launcher ----------------
extern "C" void kernel_launch(void* const* d_in, const int* in_sizes, int n_in,
                              void* d_out, int out_size, void* d_ws, size_t ws_size,
                              hipStream_t stream) {
  const float* x   = (const float*)d_in[0];
  const int*   adj = (const int*)d_in[1];
  const float* Wh  = (const float*)d_in[2];
  const float* ah  = (const float*)d_in[3];
  const float* Wo  = (const float*)d_in[4];
  const float* ao  = (const float*)d_in[5];

  char* p = (char*)d_ws;
  auto alloc = [&](size_t bytes) -> void* {
    void* r = (void*)p;
    p += (bytes + 255) & ~(size_t)255;
    return r;
  };

  unsigned*  bm     = (unsigned*)alloc((size_t)N * (N / 32) * sizeof(unsigned)); // 8 MB
  _Float16*  x16    = (_Float16*)alloc((size_t)N * NFEAT * sizeof(_Float16));    // 8 MB
  _Float16*  W16T   = (_Float16*)alloc((size_t)KCAT * NFEAT * sizeof(_Float16));
  _Float16*  Wo16T  = (_Float16*)alloc((size_t)NCLASS * KCAT * sizeof(_Float16));
  _Float16*  H16T   = (_Float16*)alloc((size_t)KCAT * N * sizeof(_Float16));     // 4 MB
  float*     f1     = (float*)alloc((size_t)NHEADS * N * sizeof(float));
  float*     f2     = (float*)alloc((size_t)NHEADS * N * sizeof(float));
  _Float16*  f2h16  = (_Float16*)alloc((size_t)NHEADS * N * sizeof(_Float16));
  float*     gmax   = (float*)alloc(NHEADS * sizeof(float));
  _Float16*  xcat16 = (_Float16*)alloc((size_t)N * KCAT * sizeof(_Float16));     // 4 MB
  _Float16*  H2T    = (_Float16*)alloc((size_t)NCLASS * N * sizeof(_Float16));
  float*     f1o    = (float*)alloc((size_t)N * sizeof(float));
  float*     f2o    = (float*)alloc((size_t)N * sizeof(float));
  _Float16*  f2o16  = (_Float16*)alloc((size_t)N * sizeof(_Float16));
  float*     gmaxo  = (float*)alloc(sizeof(float));

  k_cvt_x <<<(N * NFEAT + 255) / 256, 256, 0, stream>>>(x, x16, N * NFEAT);
  k_cvt_wh<<<(NHEADS * NFEAT * NHID + 255) / 256, 256, 0, stream>>>(Wh, W16T);
  k_cvt_wo<<<(KCAT * NCLASS + 255) / 256, 256, 0, stream>>>(Wo, Wo16T);
  k_pack  <<<dim3(N / 256, N), 256, 0, stream>>>(adj, bm);
  k_gemm1 <<<N / 16, 256, 0, stream>>>(x16, W16T, ah, H16T, f1, f2, f2h16);
  k_rowmax<<<NHEADS, 256, 0, stream>>>(f2, gmax);
  k_att1  <<<N / 32, 256, 0, stream>>>(bm, H16T, f1, f2h16, gmax, xcat16);
  k_gemm2 <<<N / 16, 32, 0, stream>>>(xcat16, Wo16T, ao, H2T, f1o, f2o, f2o16);
  k_rowmax<<<1, 256, 0, stream>>>(f2o, gmaxo);
  k_att2  <<<N / 32, 256, 0, stream>>>(bm, H2T, f1o, f2o16, gmaxo, (float*)d_out);
}
